// VisionTransformer_2937757630500
// MI455X (gfx1250) — matrix operand order repaired and checked by reading, compiled, not run on hardware
//
#include <hip/hip_runtime.h>
#include <hip/hip_bf16.h>
#include <math.h>

#define WH 62
#define WW 10
#define NTOK 621                 // WH*WW + 1
#define DIM 768
#define HEADS 12
#define HD 64
#define HIDDEN 3072
#define BATCH 16
#define MROWS (BATCH*NTOK)       // 9936
#define QKVDIM (3*DIM)           // 2304
#define ATT_SCALE 0.125f         // 64^-0.5
#define NUMREL ((2*WH-1)*(2*WW-1)+3)
#define NKT ((NTOK+31)/32)       // 20 key tiles

typedef __attribute__((ext_vector_type(8)))  __bf16 v8bf;
typedef __attribute__((ext_vector_type(16))) __bf16 v16bf;
typedef __attribute__((ext_vector_type(8)))  float  v8f;

// ---- CDNA5 async global->LDS copy (16B per lane), tracked by ASYNCcnt ----
__device__ __forceinline__ void async_copy_b128(const void* gsrc, void* lds_dst) {
  unsigned dst = (unsigned)(size_t)lds_dst;   // low 32 bits of flat addr = LDS offset
  asm volatile("global_load_async_to_lds_b128 %0, %1, off"
               :: "v"(dst), "v"(gsrc) : "memory");
}
__device__ __forceinline__ void async_wait0() {
  asm volatile("s_wait_asynccnt 0" ::: "memory");
}

// ---------------- misc kernels ----------------
__global__ __launch_bounds__(256) void f2bf_kernel(const float* __restrict__ in,
                                                   __bf16* __restrict__ out, int n) {
  int i = blockIdx.x * 256 + threadIdx.x;
  if (i < n) out[i] = (__bf16)in[i];
}

__global__ __launch_bounds__(256) void layernorm_kernel(const float* __restrict__ x,
                                                        const float* __restrict__ g,
                                                        const float* __restrict__ b,
                                                        __bf16* __restrict__ y, int M) {
  __shared__ float sh[256];
  int row = blockIdx.x;
  if (row >= M) return;
  const float* xr = x + (size_t)row * DIM;
  int tid = threadIdx.x;
  float s1 = 0.f, s2 = 0.f;
  for (int i = tid; i < DIM; i += 256) { float v = xr[i]; s1 += v; s2 += v * v; }
  sh[tid] = s1; __syncthreads();
  for (int o = 128; o > 0; o >>= 1) { if (tid < o) sh[tid] += sh[tid + o]; __syncthreads(); }
  float mean = sh[0] * (1.f / DIM); __syncthreads();
  sh[tid] = s2; __syncthreads();
  for (int o = 128; o > 0; o >>= 1) { if (tid < o) sh[tid] += sh[tid + o]; __syncthreads(); }
  float var = sh[0] * (1.f / DIM) - mean * mean;
  float inv = rsqrtf(var + 1e-5f);
  for (int i = tid; i < DIM; i += 256)
    y[(size_t)row * DIM + i] = (__bf16)((xr[i] - mean) * inv * g[i] + b[i]);
}

// ------------- WMMA GEMM: C[M,N] = A[M,K] @ W[N,K]^T -------------
// Double-buffered LDS tiles filled by async global->LDS copies.
// MODE 0: QKV (+q/v bias, scale q)            -> outB [M,2304] bf16
// MODE 1: proj (+bias, gamma1, residual)      -> outF [M,768]  f32
// MODE 2: fc1 (+bias, exact GELU)             -> outB [M,3072] bf16
// MODE 3: fc2 (+bias, gamma2, residual)       -> outF [M,768]  f32
template <int MODE, int K>
__global__ __launch_bounds__(256)
void gemm_wmma_kernel(const __bf16* __restrict__ A, const __bf16* __restrict__ W,
                      int M,
                      const float* __restrict__ bias0, const float* __restrict__ bias1,
                      const float* __restrict__ resid,
                      float* __restrict__ outF, __bf16* __restrict__ outB) {
  const int BM = 128, BN = 128, BK = 32;
  // A tile, fragment-major: [strip(8)][lane(32)][elem(16)]
  __shared__ __align__(32) __bf16 Af[2][(BM / 16) * 32 * 16];
  // W tile, column-major:   [n(128)][k(32)]  (B fragment = 16 contiguous elems)
  __shared__ __align__(32) __bf16 Bt[2][BN * 32];
  int tid = threadIdx.x, lane = tid & 31, wv = tid >> 5;
  int m0 = blockIdx.x * BM, n0 = blockIdx.y * BN;
  int colid = lane & 15, kh = (lane >> 4) & 1;

  auto fill = [&](int buf, int k0) {
    for (int ci = tid; ci < 512; ci += 256) {
      int r = ci >> 2, cb = (ci & 3) * 8;
      int gr = m0 + r; if (gr >= M) gr = M - 1;            // clamp; rows guarded at store
      int khA = (cb >> 4) & 1, jb = cb & 8;
      async_copy_b128(A + (size_t)gr * K + k0 + cb,
                      &Af[buf][(((r >> 4) * 32) + (r & 15) + 16 * khA) * 16 + jb]);
      int n = ci >> 2, kb = (ci & 3) * 8;
      async_copy_b128(W + (size_t)(n0 + n) * K + k0 + kb, &Bt[buf][n * 32 + kb]);
    }
  };

  v8f acc[8] = {};
  fill(0, 0);
  async_wait0();
  __syncthreads();
#pragma unroll 2
  for (int k0 = 0; k0 < K; k0 += BK) {
    int cur = (k0 / BK) & 1;
    if (k0 + BK < K) fill(cur ^ 1, k0 + BK);               // prefetch next tile
    v16bf a = *(const v16bf*)(&Af[cur][(wv * 32 + lane) * 16]);
#pragma unroll
    for (int t = 0; t < 8; ++t) {
      v16bf b = *(const v16bf*)(&Bt[cur][(t * 16 + colid) * 32 + kh * 16]);
      acc[t] = __builtin_amdgcn_wmma_f32_16x16x32_bf16(false, a, false, b,
                                                       (short)0, acc[t], false, false);
    }
    async_wait0();
    __syncthreads();
  }
  int rsel = kh;
#pragma unroll
  for (int t = 0; t < 8; ++t) {
#pragma unroll
    for (int r = 0; r < 8; ++r) {
      int grow = m0 + wv * 16 + r + 8 * rsel;
      if (grow >= M) continue;
      int gcol = n0 + t * 16 + colid;
      float v = acc[t][r];
      if (MODE == 0) {
        if (gcol < DIM)          v = (v + bias0[gcol]) * ATT_SCALE;
        else if (gcol >= 2 * DIM) v += bias1[gcol - 2 * DIM];
        outB[(size_t)grow * QKVDIM + gcol] = (__bf16)v;
      } else if (MODE == 1) {
        v += bias0[gcol];
        outF[(size_t)grow * DIM + gcol] = resid[(size_t)grow * DIM + gcol] + bias1[gcol] * v;
      } else if (MODE == 2) {
        v += bias0[gcol];
        v = 0.5f * v * (1.f + erff(v * 0.70710678118654752f));   // exact GELU
        outB[(size_t)grow * HIDDEN + gcol] = (__bf16)v;
      } else {
        v += bias0[gcol];
        outF[(size_t)grow * DIM + gcol] = resid[(size_t)grow * DIM + gcol] + bias1[gcol] * v;
      }
    }
  }
}

// ---------------- relative position bias (REL_IDX recomputed on device) --------
__device__ __forceinline__ float rel_bias(const float* __restrict__ tbl, int qi, int kj, int h) {
  int idx;
  if (qi == 0 && kj == 0)      idx = NUMREL - 1;
  else if (kj == 0)            idx = NUMREL - 2;
  else if (qi == 0)            idx = NUMREL - 3;
  else {
    int ih = (qi - 1) / WW, iw = (qi - 1) % WW;
    int kh2 = (kj - 1) / WW, kw = (kj - 1) % WW;
    idx = (ih - kh2 + WH - 1) * (2 * WW - 1) + (iw - kw + WW - 1);
  }
  return tbl[idx * HEADS + h];
}

// -------- fused flash attention: 4 waves/block share K/V stage, 1 q-tile/wave --------
__global__ __launch_bounds__(128)
void attention_kernel(const __bf16* __restrict__ qkv, const float* __restrict__ rel_table,
                      __bf16* __restrict__ out) {
  __shared__ __align__(32) __bf16 Qf[4][2][512];   // per wave: 2 A-fragments (d 0..31 / 32..63)
  __shared__ __align__(32) __bf16 Ks[32 * 80];     // row-major, 160B row stride (32B aligned)
  __shared__ __align__(32) __bf16 Vf[4 * 32 * 16]; // B-fragment-major: [t][lane][elem]
  __shared__ __align__(32) __bf16 Pf[4][512];      // per wave: P A-fragment
  int tid = threadIdx.x, lane = tid & 31, wv = tid >> 5;
  int b = blockIdx.x / HEADS, h = blockIdx.x % HEADS;
  int q0 = (blockIdx.y * 4 + wv) * 16;
  const __bf16* base = qkv + (size_t)b * NTOK * QKVDIM;
  int colid = lane & 15, rsel = lane >> 4;

  // Q fill (per-wave) via async copies -> fragment-major
  for (int ci = lane; ci < 128; ci += 32) {
    int r = ci >> 3, cb = (ci & 7) * 8;
    int qr = q0 + r; if (qr >= NTOK) qr = NTOK - 1;        // clamp; rows guarded at store
    int dk = cb >> 5, c32 = cb & 31;
    int khq = (c32 >> 4) & 1, jb = c32 & 8;
    async_copy_b128(base + (size_t)qr * QKVDIM + h * HD + cb,
                    &Qf[wv][dk][(r + 16 * khq) * 16 + jb]);
  }
  async_wait0();
  __syncthreads();
  v16bf aq0 = *(const v16bf*)(&Qf[wv][0][lane * 16]);
  v16bf aq1 = *(const v16bf*)(&Qf[wv][1][lane * 16]);

  v8f oacc[4] = {};
  float mrow[8], lrow[8];
#pragma unroll
  for (int r = 0; r < 8; ++r) { mrow[r] = -1e30f; lrow[r] = 0.f; }

  for (int kt = 0; kt < NKT; ++kt) {
    int k0 = kt * 32;
    __syncthreads();
    // K fill via async copies, row-major (K^T fragments are contiguous reads)
    for (int ci = tid; ci < 256; ci += 128) {
      int r = ci >> 3, cb = (ci & 7) * 8;
      int kr = k0 + r; if (kr >= NTOK) kr = NTOK - 1;      // clamp; masked to -inf below
      async_copy_b128(base + (size_t)kr * QKVDIM + DIM + h * HD + cb, Ks + r * 80 + cb);
    }
    // V fill: vector loads, scattered into B-fragment-major order
    for (int ci = tid; ci < 256; ci += 128) {
      int kk = ci >> 3, d0 = (ci & 7) * 8, kr = k0 + kk;
      v8bf v = {};
      if (kr < NTOK) v = *(const v8bf*)(base + (size_t)kr * QKVDIM + 2 * DIM + h * HD + d0);
      int t = d0 >> 4, j = kk & 15, lbase = (d0 & 15) + 16 * (kk >> 4);
#pragma unroll
      for (int e = 0; e < 8; ++e)
        Vf[(t * 32 + lbase + e) * 16 + j] = v[e];
    }
    async_wait0();
    __syncthreads();

    v8f sc[2] = {};
#pragma unroll
    for (int t = 0; t < 2; ++t) {
      v16bf b0 = *(const v16bf*)(Ks + (t * 16 + colid) * 80 + rsel * 16);      // d 0..31
      sc[t] = __builtin_amdgcn_wmma_f32_16x16x32_bf16(false, aq0, false, b0, (short)0, sc[t], false, false);
      v16bf b1 = *(const v16bf*)(Ks + (t * 16 + colid) * 80 + 32 + rsel * 16); // d 32..63
      sc[t] = __builtin_amdgcn_wmma_f32_16x16x32_bf16(false, aq1, false, b1, (short)0, sc[t], false, false);
    }
#pragma unroll
    for (int t = 0; t < 2; ++t)
#pragma unroll
      for (int r = 0; r < 8; ++r) {
        int qi = q0 + r + 8 * rsel, kj = k0 + t * 16 + colid;
        float s = sc[t][r];
        sc[t][r] = (qi < NTOK && kj < NTOK) ? s + rel_bias(rel_table, qi, kj, h) : -1e30f;
      }
    // online softmax: each row lives across the 16 lanes of a half-wave
#pragma unroll
    for (int r = 0; r < 8; ++r) {
      float v = fmaxf(sc[0][r], sc[1][r]);
#pragma unroll
      for (int o = 1; o < 16; o <<= 1) v = fmaxf(v, __shfl_xor(v, o, 32));
      float mn = fmaxf(mrow[r], v);
      float corr = __expf(mrow[r] - mn);
      mrow[r] = mn;
      float p0 = __expf(sc[0][r] - mn), p1 = __expf(sc[1][r] - mn);
      sc[0][r] = p0; sc[1][r] = p1;
      float ps = p0 + p1;
#pragma unroll
      for (int o = 1; o < 16; o <<= 1) ps += __shfl_xor(ps, o, 32);
      lrow[r] = lrow[r] * corr + ps;
#pragma unroll
      for (int t = 0; t < 4; ++t) oacc[t][r] *= corr;
    }
    // P -> fragment-major LDS (per wave), then P @ V
#pragma unroll
    for (int t = 0; t < 2; ++t)
#pragma unroll
      for (int r = 0; r < 8; ++r) {
        int m = r + 8 * rsel;
        Pf[wv][((m & 15) + 16 * t) * 16 + colid] = (__bf16)sc[t][r];
      }
    v16bf ap = *(const v16bf*)(&Pf[wv][lane * 16]);
#pragma unroll
    for (int t = 0; t < 4; ++t) {
      v16bf bv = *(const v16bf*)(Vf + (t * 32 + lane) * 16);
      oacc[t] = __builtin_amdgcn_wmma_f32_16x16x32_bf16(false, ap, false, bv, (short)0, oacc[t], false, false);
    }
  }
  __bf16* ob = out + (size_t)b * NTOK * DIM + h * HD;
#pragma unroll
  for (int r = 0; r < 8; ++r) {
    int qi = q0 + r + 8 * rsel;
    if (qi >= NTOK) continue;
    float inv = 1.f / lrow[r];
#pragma unroll
    for (int t = 0; t < 4; ++t)
      ob[(size_t)qi * DIM + t * 16 + colid] = (__bf16)(oacc[t][r] * inv);
  }
}

// ---------------- launch ----------------
extern "C" void kernel_launch(void* const* d_in, const int* in_sizes, int n_in,
                              void* d_out, int out_size, void* d_ws, size_t ws_size,
                              hipStream_t stream) {
  const float* x        = (const float*)d_in[0];
  const float* ln1_g    = (const float*)d_in[1];
  const float* ln1_b    = (const float*)d_in[2];
  const float* w_qkv    = (const float*)d_in[3];
  const float* q_bias   = (const float*)d_in[4];
  const float* v_bias   = (const float*)d_in[5];
  const float* rel_tab  = (const float*)d_in[6];
  const float* proj_w   = (const float*)d_in[7];
  const float* proj_b   = (const float*)d_in[8];
  const float* gamma1   = (const float*)d_in[9];
  const float* gamma2   = (const float*)d_in[10];
  const float* ln2_g    = (const float*)d_in[11];
  const float* ln2_b    = (const float*)d_in[12];
  const float* fc1_w    = (const float*)d_in[13];
  const float* fc1_b    = (const float*)d_in[14];
  const float* fc2_w    = (const float*)d_in[15];
  const float* fc2_b    = (const float*)d_in[16];
  float* outp = (float*)d_out;

  char* ws = (char*)d_ws;
  size_t off = 0;
  auto alloc = [&](size_t bytes) { char* p = ws + off; off = (off + bytes + 255) & ~(size_t)255; return p; };
  __bf16* wqkv_bf = (__bf16*)alloc((size_t)QKVDIM * DIM * 2);
  __bf16* wprj_bf = (__bf16*)alloc((size_t)DIM * DIM * 2);
  __bf16* wfc1_bf = (__bf16*)alloc((size_t)HIDDEN * DIM * 2);
  __bf16* wfc2_bf = (__bf16*)alloc((size_t)DIM * HIDDEN * 2);
  __bf16* actA    = (__bf16*)alloc((size_t)MROWS * HIDDEN * 2);  // xn, later h
  __bf16* qkv     = (__bf16*)alloc((size_t)MROWS * QKVDIM * 2);
  __bf16* actB    = (__bf16*)alloc((size_t)MROWS * DIM * 2);     // attn-out, later yn
  float*  x1      = (float*)alloc((size_t)MROWS * DIM * 4);

  auto cvt = [&](const float* src, __bf16* dst, int n) {
    f2bf_kernel<<<(n + 255) / 256, 256, 0, stream>>>(src, dst, n);
  };
  cvt(w_qkv, wqkv_bf, QKVDIM * DIM);
  cvt(proj_w, wprj_bf, DIM * DIM);
  cvt(fc1_w, wfc1_bf, HIDDEN * DIM);
  cvt(fc2_w, wfc2_bf, DIM * HIDDEN);

  // 1) LN1 -> xn (bf16)
  layernorm_kernel<<<MROWS, 256, 0, stream>>>(x, ln1_g, ln1_b, actA, MROWS);
  // 2) QKV GEMM
  gemm_wmma_kernel<0, DIM><<<dim3((MROWS + 127) / 128, QKVDIM / 128), 256, 0, stream>>>(
      actA, wqkv_bf, MROWS, q_bias, v_bias, nullptr, nullptr, qkv);
  // 3) fused flash attention -> actB (bf16)
  attention_kernel<<<dim3(BATCH * HEADS, ((NTOK + 15) / 16 + 3) / 4), 128, 0, stream>>>(
      qkv, rel_tab, actB);
  // 4) proj GEMM + gamma1 residual -> x1 (f32)
  gemm_wmma_kernel<1, DIM><<<dim3((MROWS + 127) / 128, DIM / 128), 256, 0, stream>>>(
      actB, wprj_bf, MROWS, proj_b, gamma1, x, x1, nullptr);
  // 5) LN2 -> yn (bf16, reuse actB)
  layernorm_kernel<<<MROWS, 256, 0, stream>>>(x1, ln2_g, ln2_b, actB, MROWS);
  // 6) fc1 GEMM + GELU -> h (bf16, reuse actA)
  gemm_wmma_kernel<2, DIM><<<dim3((MROWS + 127) / 128, HIDDEN / 128), 256, 0, stream>>>(
      actB, wfc1_bf, MROWS, fc1_b, nullptr, nullptr, nullptr, actA);
  // 7) fc2 GEMM + gamma2 residual -> d_out (f32)
  gemm_wmma_kernel<3, HIDDEN><<<dim3((MROWS + 127) / 128, DIM / 128), 256, 0, stream>>>(
      actA, wfc2_bf, MROWS, fc2_b, gamma2, x1, outp, nullptr);
  (void)in_sizes; (void)n_in; (void)out_size; (void)ws_size;
}
